// Conv2dMod_2972117369293
// MI455X (gfx1250) — matrix operand order repaired and checked by reading, not compiled, not run on hardware
//
#include <hip/hip_runtime.h>

typedef __attribute__((ext_vector_type(16))) __bf16 v16bf;
typedef __attribute__((ext_vector_type(8)))  __bf16 v8bf;
typedef __attribute__((ext_vector_type(8)))  float  v8f;
typedef __attribute__((ext_vector_type(4)))  int    v4i;

#define CIN    512
#define COUT   512
#define NBATCH 8
#define HWSZ   1024     // 32*32
#define KTOT   4608     // CIN*9
#define HP     34       // padded spatial
#define LDT    40       // LDS row stride (elements): 32 + 8 pad
#define BM     128
#define BN     128

#if __has_builtin(__builtin_amdgcn_global_load_async_to_lds_b128)
#define HAVE_ASYNC_LDS 1
typedef __attribute__((address_space(1))) v4i gv4i;   // global 16B vector
typedef __attribute__((address_space(3))) v4i sv4i;   // LDS 16B vector
#define ASYNC_B128(gp, lp, off) \
    __builtin_amdgcn_global_load_async_to_lds_b128((gv4i*)(unsigned short*)(gp), \
                                                   (sv4i*)(unsigned short*)(lp), (off), 0)
#else
#define HAVE_ASYNC_LDS 0
#endif

__device__ __forceinline__ void wait_async0() {
#if __has_builtin(__builtin_amdgcn_s_wait_asynccnt)
    __builtin_amdgcn_s_wait_asynccnt(0);
#else
    asm volatile("s_wait_asynccnt 0x0" ::: "memory");
#endif
}

__device__ __forceinline__ unsigned short f2bf(float f) {
    unsigned u = __float_as_uint(f);
    return (unsigned short)((u + 0x7fffu + ((u >> 16) & 1u)) >> 16);  // RNE
}

// One block per output channel o: wsq[o] = sum W[o]^2, then scale[n,o] for all n.
__global__ __launch_bounds__(256) void scale_kernel(const float* __restrict__ y,
                                                    const float* __restrict__ w,
                                                    float* __restrict__ scale) {
    __shared__ float red[256];
    const int o = blockIdx.x;
    const float* wr = w + (size_t)o * KTOT;
    float acc = 0.f;
    for (int t = threadIdx.x; t < KTOT; t += 256) { float v = wr[t]; acc += v * v; }
    red[threadIdx.x] = acc;
    __syncthreads();
    for (int s = 128; s > 0; s >>= 1) {
        if (threadIdx.x < s) red[threadIdx.x] += red[threadIdx.x + s];
        __syncthreads();
    }
    if (threadIdx.x < NBATCH) {
        float wsq = red[0];
        float s = 1.0f + y[threadIdx.x * COUT + o];
        scale[threadIdx.x * COUT + o] = s * rsqrtf(wsq * s * s + 1e-8f);
    }
}

// W[o][c][kh][kw] f32  ->  Wb[o][khkw][c] bf16   (K-contiguous rows for WMMA A)
__global__ __launch_bounds__(256) void wconv_kernel(const float* __restrict__ w,
                                                    unsigned short* __restrict__ wb) {
    const int idx = blockIdx.x * 256 + threadIdx.x;   // exactly 512*9*512 = 9216*256
    const int c  = idx & 511;
    const int kk = (idx >> 9) % 9;
    const int o  = idx / (512 * 9);
    wb[idx] = f2bf(w[((size_t)(o * 512 + c)) * 9 + kk]);
}

// x NCHW f32 -> xb[n][hp][wp][c] bf16 with 1-px replicate border (NHWC, C-contiguous)
__global__ __launch_bounds__(256) void xpad_kernel(const float* __restrict__ x,
                                                   unsigned short* __restrict__ xb) {
    const long idx = (long)blockIdx.x * 256 + threadIdx.x;  // exactly 8*34*34*512 = 18496*256
    const int c  = (int)(idx & 511);
    const int wp = (int)((idx >> 9) % HP);
    const int hp = (int)((idx / (512 * HP)) % HP);
    const int n  = (int)(idx / (512 * HP * HP));
    const int h  = min(max(hp - 1, 0), 31);
    const int w  = min(max(wp - 1, 0), 31);
    xb[idx] = f2bf(x[(((size_t)(n * CIN + c)) * 32 + h) * 32 + w]);
}

__device__ __forceinline__ v16bf cat16(v8bf lo, v8bf hi) {
    return __builtin_shufflevector(lo, hi, 0,1,2,3,4,5,6,7,8,9,10,11,12,13,14,15);
}

// Implicit GEMM: 128x128 output tile per block; 32(M)x64(N) per wave; K-steps of 32.
__global__ __launch_bounds__(256) void conv_wmma_kernel(const unsigned short* __restrict__ wb,
                                                        const unsigned short* __restrict__ xb,
                                                        const float* __restrict__ scale,
                                                        float* __restrict__ out) {
    __shared__ __align__(16) unsigned short lA[2][BM * LDT];
    __shared__ __align__(16) unsigned short lB[2][BN * LDT];

    const int tid   = threadIdx.x;
    const int lane  = tid & 31;
    const int wave  = tid >> 5;          // 8 waves
    const int mwave = wave >> 1;         // 0..3 : 32-row slab
    const int nwave = wave & 1;          // 0..1 : 64-col slab
    const int mtile = blockIdx.y * BM;   // 4 tiles over COUT
    const int ntile = blockIdx.x * BN;   // 64 tiles over N*H*W

    // ---- global -> LDS assignments: thread t = (row r, 32B half) ----
    const int r  = tid >> 1;             // 0..127
    const int ho = (tid & 1) * 16;       // element offset 0 or 16 (two b128 each)
    const unsigned short* aSrc = wb + (size_t)(mtile + r) * KTOT + ho;
    const int col = ntile + r;
    const int n   = col >> 10;           // sample (uniform per block: 128 | 1024)
    const int hw  = col & 1023;
    const int h   = hw >> 5;
    const int w   = hw & 31;
    const unsigned short* bSrc = xb + ((size_t)((n * HP + h) * HP + w)) * CIN + ho;
    const int aOff = r * LDT + ho;       // LDS element offset (same for A and B tiles)

    // ---- per-lane fragment addresses (ISA 16-bit A 16x32 / B 32x16 layouts) ----
    const int l15 = lane & 15;
    const int kh0 = (lane >> 4) << 3;    // A: lanes 16-31 hold K=8..15 / 24..31
    const int kb0 = (lane >> 4) << 3;    // B: lanes 16-31 hold K=8..15 / 24..31
    const int aRow0 = (mwave * 32 + l15) * LDT + kh0;       // mi adds 16*LDT
    const int bCol0 = (nwave * 64 + l15) * LDT + kb0;       // nj adds 16*LDT

    v8f acc[2][4];
#pragma unroll
    for (int mi = 0; mi < 2; ++mi)
#pragma unroll
        for (int nj = 0; nj < 4; ++nj) acc[mi][nj] = (v8f){};

    // ---- prologue: tile 0 -> buffer 0 ----
#if HAVE_ASYNC_LDS
    ASYNC_B128(aSrc, &lA[0][aOff], 0);
    ASYNC_B128(aSrc, &lA[0][aOff], 16);
    ASYNC_B128(bSrc, &lB[0][aOff], 0);
    ASYNC_B128(bSrc, &lB[0][aOff], 16);
    wait_async0();
#else
    {
        uint4 a0 = *(const uint4*)(aSrc),      a1 = *(const uint4*)(aSrc + 8);
        uint4 b0 = *(const uint4*)(bSrc),      b1 = *(const uint4*)(bSrc + 8);
        *(uint4*)&lA[0][aOff]     = a0;  *(uint4*)&lA[0][aOff + 8] = a1;
        *(uint4*)&lB[0][aOff]     = b0;  *(uint4*)&lB[0][aOff + 8] = b1;
    }
#endif
    __syncthreads();

    for (int ks = 0, it = 0; ks < KTOT; ks += 32, ++it) {
        const int cur = it & 1, nxt = cur ^ 1;
        const int ksn = ks + 32;
        const bool more = (ksn < KTOT);          // uniform

        // element offset of next B chunk inside padded NHWC image
        int bOffN = 0;
        if (more) {
            const int kk = ksn >> 9;             // which (kh,kw)
            const int c0 = ksn & 511;
            bOffN = ((kk / 3) * HP + (kk % 3)) * CIN + c0;
        }

#if HAVE_ASYNC_LDS
        if (more) {
            ASYNC_B128(aSrc + ksn,   &lA[nxt][aOff], 0);
            ASYNC_B128(aSrc + ksn,   &lA[nxt][aOff], 16);
            ASYNC_B128(bSrc + bOffN, &lB[nxt][aOff], 0);
            ASYNC_B128(bSrc + bOffN, &lB[nxt][aOff], 16);
        }
#else
        uint4 a0v, a1v, b0v, b1v;
        if (more) {
            a0v = *(const uint4*)(aSrc + ksn);   a1v = *(const uint4*)(aSrc + ksn + 8);
            b0v = *(const uint4*)(bSrc + bOffN); b1v = *(const uint4*)(bSrc + bOffN + 8);
        }
#endif

        // ---- compute from current buffer: 2 A-frags x 4 B-frags = 8 WMMAs ----
        const unsigned short* Ab = &lA[cur][aRow0];
        const unsigned short* Bb = &lB[cur][bCol0];

        v16bf af[2], bf[4];
#pragma unroll
        for (int mi = 0; mi < 2; ++mi) {
            const unsigned short* p = Ab + mi * 16 * LDT;
            af[mi] = cat16(*(const v8bf*)(p), *(const v8bf*)(p + 16));
        }
#pragma unroll
        for (int nj = 0; nj < 4; ++nj) {
            const unsigned short* p = Bb + nj * 16 * LDT;
            bf[nj] = cat16(*(const v8bf*)(p), *(const v8bf*)(p + 16));
        }
#pragma unroll
        for (int mi = 0; mi < 2; ++mi)
#pragma unroll
            for (int nj = 0; nj < 4; ++nj)
                acc[mi][nj] = __builtin_amdgcn_wmma_f32_16x16x32_bf16(
                    false, af[mi], false, bf[nj], (short)0, acc[mi][nj], false, false);

#if HAVE_ASYNC_LDS
        wait_async0();                 // next tile landed in LDS
#else
        if (more) {
            *(uint4*)&lA[nxt][aOff]     = a0v;  *(uint4*)&lA[nxt][aOff + 8] = a1v;
            *(uint4*)&lB[nxt][aOff]     = b0v;  *(uint4*)&lB[nxt][aOff + 8] = b1v;
        }
#endif
        __syncthreads();
    }

    // ---- epilogue: scale by scale[n,m] and store NCHW f32 ----
    const int nOut  = ntile >> 10;
    const int hwB   = (ntile & 1023) + nwave * 64 + l15;
    const int mrow0 = mtile + mwave * 32 + ((lane >> 4) << 3);   // C/D layout halves
#pragma unroll
    for (int mi = 0; mi < 2; ++mi) {
#pragma unroll
        for (int rr = 0; rr < 8; ++rr) {
            const int m = mrow0 + mi * 16 + rr;
            const float s = scale[nOut * COUT + m];
            const size_t base = ((size_t)(nOut * COUT + m)) * HWSZ + hwB;
#pragma unroll
            for (int nj = 0; nj < 4; ++nj)
                out[base + nj * 16] = acc[mi][nj][rr] * s;
        }
    }
}

extern "C" void kernel_launch(void* const* d_in, const int* in_sizes, int n_in,
                              void* d_out, int out_size, void* d_ws, size_t ws_size,
                              hipStream_t stream) {
    const float* x = (const float*)d_in[0];   // [8,512,32,32]
    const float* y = (const float*)d_in[1];   // [8,512]
    const float* w = (const float*)d_in[2];   // [512,512,3,3]
    float* out = (float*)d_out;               // [8,512,32,32]

    char* ws = (char*)d_ws;
    float* scale       = (float*)ws;                              // 16384 B
    unsigned short* wb = (unsigned short*)(ws + 16384);           // 4,718,592 B
    unsigned short* xb = (unsigned short*)(ws + 16384 + 4718592); // 9,469,952 B
    // total ws usage ~13.6 MB

    scale_kernel<<<COUT, 256, 0, stream>>>(y, w, scale);
    wconv_kernel<<<(COUT * 9 * CIN) / 256, 256, 0, stream>>>(w, wb);
    xpad_kernel<<<(NBATCH * HP * HP * CIN) / 256, 256, 0, stream>>>(x, xb);

    dim3 grid(64, 4);   // 64 N-tiles x 4 M-tiles, 128x128 outputs each
    conv_wmma_kernel<<<grid, 256, 0, stream>>>(wb, xb, scale, out);
}
